// MultiHeadAttention_2714419331921
// MI455X (gfx1250) — matrix operand order repaired and checked by reading, not compiled, not run on hardware
//
#include <hip/hip_runtime.h>
#include <hip/hip_bf16.h>

typedef unsigned short u16;
typedef __attribute__((ext_vector_type(16))) __bf16 v16bf;
typedef __attribute__((ext_vector_type(8)))  float   v8f;
typedef __attribute__((ext_vector_type(4)))  int     v4i;

#define AS_GLOBAL __attribute__((address_space(1)))
#define AS_LDS    __attribute__((address_space(3)))

// POD 16-byte chunk so it can live in a union with the bf16 vector.
struct alignas(16) U4 { unsigned int w[4]; };
union Frag16 { U4 q[2]; v16bf v; };   // 32 bytes = 8 VGPRs = one WMMA A/B operand

#define D_MODEL 2048
#define T_SEQ   2048
#define NBATCH  4
#define NH      16
#define DH      128
#define DHALF   64

__device__ __forceinline__ u16 f2bf(float f) {
  unsigned int u = __float_as_uint(f);
  u += 0x7FFFu + ((u >> 16) & 1u);          // round-to-nearest-even
  return (u16)(u >> 16);
}

// xor-lane shuffle via ds_swizzle (immediate encoding: and=0x1f, or=0, xor=m)
#define SWZ_XOR(v, m) \
  __int_as_float(__builtin_amdgcn_ds_swizzle(__float_as_int(v), (0x1f | ((m) << 10))))

// ---------------------------------------------------------------------------
// CDNA5 async global->LDS copy (16 B per lane). ASYNCcnt-tracked; no VGPR
// data staging, so the double buffers cost zero registers.
// Builtin signature (from hipcc diagnostic): (int4 AS1*, int4 AS3*, imm, cpol)
// ---------------------------------------------------------------------------
__device__ __forceinline__ void cp_async16(u16* l, const u16* g) {
#if __has_builtin(__builtin_amdgcn_global_load_async_to_lds_b128)
  __builtin_amdgcn_global_load_async_to_lds_b128(
      (AS_GLOBAL v4i*)g, (AS_LDS v4i*)l, 0, 0);
#else
  unsigned loff = (unsigned)(size_t)(AS_LDS const u16*)l;
  asm volatile("global_load_async_to_lds_b128 %0, %1, off"
               :: "v"(loff), "v"(g) : "memory");
#endif
}

__device__ __forceinline__ void wait_async0() {
#if __has_builtin(__builtin_amdgcn_s_wait_asynccnt)
  __builtin_amdgcn_s_wait_asynccnt(0);
#else
  asm volatile("s_wait_asynccnt 0" ::: "memory");
#endif
}

// ---------------------------------------------------------------------------
// Kernel 1: x fp32 -> bf16 (row-major [8192][2048])
// ---------------------------------------------------------------------------
__global__ void convx_kernel(const float* __restrict__ x, u16* __restrict__ xb, int n4) {
  int i = blockIdx.x * blockDim.x + threadIdx.x;
  if (i >= n4) return;
  float4 f = ((const float4*)x)[i];
  unsigned int lo = (unsigned int)f2bf(f.x) | ((unsigned int)f2bf(f.y) << 16);
  unsigned int hi = (unsigned int)f2bf(f.z) | ((unsigned int)f2bf(f.w) << 16);
  ((unsigned int*)xb)[2 * i]     = lo;
  ((unsigned int*)xb)[2 * i + 1] = hi;
}

// ---------------------------------------------------------------------------
// Kernel 2: W fp32 [d][e] -> bf16 transposed [e][d] (tiled LDS transpose)
// ---------------------------------------------------------------------------
__global__ void wtrans_kernel(const float* __restrict__ W, u16* __restrict__ Wt) {
  __shared__ float tile[32][33];
  int e  = blockIdx.x * 32 + threadIdx.x;
  int d0 = blockIdx.y * 32;
#pragma unroll
  for (int j = 0; j < 4; ++j)
    tile[threadIdx.y + 8 * j][threadIdx.x] = W[(size_t)(d0 + threadIdx.y + 8 * j) * D_MODEL + e];
  __syncthreads();
  int d  = d0 + threadIdx.x;
  int e0 = blockIdx.x * 32;
#pragma unroll
  for (int j = 0; j < 4; ++j)
    Wt[(size_t)(e0 + threadIdx.y + 8 * j) * D_MODEL + d] = f2bf(tile[threadIdx.x][threadIdx.y + 8 * j]);
}

// ---------------------------------------------------------------------------
// Kernel 3: GEMM C[64x128] = A[8192x2048](bf16) * W (via Wt [e][d] bf16)
// Async-copy double-buffered LDS, one barrier per K-step.
// MODE 0: RoPE epilogue -> dstB = Q or K, layout [b][h][t][128] bf16
// MODE 1: transpose epilogue -> dstB = V^T, layout [b][h][d][T] bf16
// MODE 2: fp32 epilogue -> dstF row-major [8192][2048]
// grid (128, 16), block 128 (4 waves x 16 rows each)
// ---------------------------------------------------------------------------
template <int MODE>
__global__ void __launch_bounds__(128, 1)
proj_kernel(const u16* __restrict__ A, const u16* __restrict__ Bt,
            const float* __restrict__ cosT, const float* __restrict__ sinT,
            u16* __restrict__ dstB, float* __restrict__ dstF) {
  __shared__ u16  lA[2][64 * 32];    // [m][k] double-buffered (8 KB)
  __shared__ u16  lB[2][128 * 32];   // [n][k] double-buffered (16 KB)
  __shared__ float lC[64 * 128];     // epilogue staging (32 KB)

  const int tid  = threadIdx.x;
  const int lane = tid & 31;
  const int wave = tid >> 5;
  const int l15  = lane & 15;
  const int g    = lane >> 4;

  const int m0 = blockIdx.x * 64;
  const int nb = blockIdx.y;
  const int n0 = nb * 128;

  // Per-thread staging coordinates (2 A-chunks, 4 B-chunks of 16B each).
  int rA[2], oA[2], rB[4], oB[4];
#pragma unroll
  for (int i = 0; i < 2; ++i) { int c = tid + i * 128; rA[i] = c >> 2; oA[i] = (c & 3) * 8; }
#pragma unroll
  for (int i = 0; i < 4; ++i) { int c = tid + i * 128; rB[i] = c >> 2; oB[i] = (c & 3) * 8; }

  v8f acc[8] = {};

  // preload K-tile 0 (async DMA into LDS)
#pragma unroll
  for (int i = 0; i < 2; ++i)
    cp_async16(&lA[0][rA[i] * 32 + oA[i]], &A[(size_t)(m0 + rA[i]) * D_MODEL + oA[i]]);
#pragma unroll
  for (int i = 0; i < 4; ++i)
    cp_async16(&lB[0][rB[i] * 32 + oB[i]], &Bt[(size_t)(n0 + rB[i]) * D_MODEL + oB[i]]);
  wait_async0();
  __syncthreads();

  const int NK = D_MODEL / 32;
  const int arow = wave * 16 + l15;
  for (int kt = 0; kt < NK; ++kt) {
    const int cur = kt & 1;
    const int nxt = cur ^ 1;
    const bool more = (kt + 1 < NK);
    if (more) {                       // async-stage next tile; hidden by WMMAs
      const int k0 = (kt + 1) * 32;
#pragma unroll
      for (int i = 0; i < 2; ++i)
        cp_async16(&lA[nxt][rA[i] * 32 + oA[i]], &A[(size_t)(m0 + rA[i]) * D_MODEL + k0 + oA[i]]);
#pragma unroll
      for (int i = 0; i < 4; ++i)
        cp_async16(&lB[nxt][rB[i] * 32 + oB[i]], &Bt[(size_t)(n0 + rB[i]) * D_MODEL + k0 + oB[i]]);
    }

    Frag16 a;   // A 16x32: lane=M, VGPR0-3: K=g*8..+7, VGPR4-7: K=16+g*8..+7
    a.q[0] = *(const U4*)&lA[cur][arow * 32 + g * 8];
    a.q[1] = *(const U4*)&lA[cur][arow * 32 + 16 + g * 8];
#pragma unroll
    for (int nf = 0; nf < 8; ++nf) {
      Frag16 b; // B 32x16: lane=N, VGPR0-3: K=g*8..+7, VGPR4-7: K=16+g*8..+7
      const int brow = nf * 16 + l15;
      b.q[0] = *(const U4*)&lB[cur][brow * 32 + g * 8];
      b.q[1] = *(const U4*)&lB[cur][brow * 32 + 16 + g * 8];
      acc[nf] = __builtin_amdgcn_wmma_f32_16x16x32_bf16(
          false, a.v, false, b.v, (short)0, acc[nf], false, false);
    }

    wait_async0();
    __syncthreads();
  }

  // C layout: VGPR r <-> row M = r + 8*g, col N = l15
#pragma unroll
  for (int nf = 0; nf < 8; ++nf)
#pragma unroll
    for (int r = 0; r < 8; ++r)
      lC[(wave * 16 + r + 8 * g) * 128 + nf * 16 + l15] = acc[nf][r];
  __syncthreads();

  const int h = nb;
  if (MODE == 0) {                       // RoPE -> [b][h][t][128] bf16
    const int c = tid;
    for (int row = 0; row < 64; ++row) {
      int mg = m0 + row;
      int b = mg >> 11, t = mg & (T_SEQ - 1);
      float v;
      if (c < DHALF) {
        float x1 = lC[row * 128 + c], x2 = lC[row * 128 + c + DHALF];
        v = x1 * cosT[t * DHALF + c] - x2 * sinT[t * DHALF + c];
      } else {
        int cc = c - DHALF;
        float x2 = lC[row * 128 + c], x1 = lC[row * 128 + cc];
        v = x2 * cosT[t * DHALF + cc] + x1 * sinT[t * DHALF + cc];
      }
      dstB[(((size_t)b * NH + h) * T_SEQ + t) * DH + c] = f2bf(v);
    }
  } else if (MODE == 1) {                // V transposed -> [b][h][d][T] bf16
    for (int i = 0; i < 64; ++i) {
      int idx = i * 128 + tid;
      int row = idx & 63, c = idx >> 6;  // consecutive tid -> consecutive t
      int mg = m0 + row;
      int b = mg >> 11, t = mg & (T_SEQ - 1);
      dstB[(((size_t)b * NH + h) * DH + c) * T_SEQ + t] = f2bf(lC[row * 128 + c]);
    }
  } else {                               // fp32 output
    const int c = tid;
    for (int row = 0; row < 64; ++row)
      dstF[(size_t)(m0 + row) * D_MODEL + n0 + c] = lC[row * 128 + c];
  }
}

// ---------------------------------------------------------------------------
// Kernel 4: causal flash attention. grid (T/64, NH, B), block 128.
// Q,K: [b][h][t][128] bf16 ; Vt: [b][h][d][T] bf16 ; O: [b*t][2048] bf16
// K/V tiles double-buffered via async DMA (no staging registers).
// Softmax row-sum rides the matrix pipe as a 9th accumulator (P @ ones);
// only the row-max uses cross-lane ds_swizzle reductions.
// ---------------------------------------------------------------------------
__global__ void __launch_bounds__(128, 1)
flash_kernel(const u16* __restrict__ Q, const u16* __restrict__ Kb,
             const u16* __restrict__ Vt, u16* __restrict__ O) {
  __shared__ u16 lQ[64 * 128];       // Q tile, reused for O staging (16 KB)
  __shared__ u16 lK[2][32 * 128];    // [key][d] double-buffered (16 KB)
  __shared__ u16 lV[2][128 * 32];    // [d][t]   double-buffered (16 KB)
  __shared__ u16 lP[4][16 * 32];     // per-wave P staging (4 KB)

  const int tid = threadIdx.x, lane = tid & 31, wave = tid >> 5;
  const int l15 = lane & 15, g = lane >> 4;
  const int b = blockIdx.z, h = blockIdx.y;
  const int qm0 = blockIdx.x * 64;
  const size_t headQK = ((size_t)b * NH + h) * T_SEQ * DH;
  const size_t headV  = ((size_t)b * NH + h) * DH * T_SEQ;

  // Per-thread staging coordinates (16B chunks).
  int rK[4], oK[4], rV[4], oV[4];
#pragma unroll
  for (int i = 0; i < 4; ++i) {
    int c = i * 128 + tid;
    rK[i] = c >> 4; oK[i] = (c & 15) * 8;   // K tile 32x128
    rV[i] = c >> 2; oV[i] = (c & 3) * 8;    // V^T tile 128x32
  }

  // async-stage Q tile 64x128 + key/value tile 0
#pragma unroll
  for (int i = 0; i < 8; ++i) {
    int c = i * 128 + tid;
    int row = c >> 4, off = (c & 15) * 8;
    cp_async16(&lQ[row * 128 + off], &Q[headQK + (size_t)(qm0 + row) * DH + off]);
  }
#pragma unroll
  for (int i = 0; i < 4; ++i)
    cp_async16(&lK[0][rK[i] * 128 + oK[i]], &Kb[headQK + (size_t)rK[i] * DH + oK[i]]);
#pragma unroll
  for (int i = 0; i < 4; ++i)
    cp_async16(&lV[0][rV[i] * 32 + oV[i]],  &Vt[headV + (size_t)rV[i] * T_SEQ + oV[i]]);
  wait_async0();
  __syncthreads();

  Frag16 qa[4];
  {
    const int row = wave * 16 + l15;
#pragma unroll
    for (int kd = 0; kd < 4; ++kd) {
      qa[kd].q[0] = *(const U4*)&lQ[row * 128 + kd * 32 + g * 8];
      qa[kd].q[1] = *(const U4*)&lQ[row * 128 + kd * 32 + 16 + g * 8];
    }
  }

  Frag16 ones;                         // all-1.0 bf16 B fragment (for row sums)
#pragma unroll
  for (int i = 0; i < 4; ++i) { ones.q[0].w[i] = 0x3F803F80u; ones.q[1].w[i] = 0x3F803F80u; }

  v8f o[8] = {};
  v8f lsum = {};                       // row-sum accumulator (9th C fragment)
  float m[8];
#pragma unroll
  for (int r = 0; r < 8; ++r) m[r] = -1e30f;

  const int nkb = qm0 / 32 + 2;                          // causal: keys 0 .. qm0+63
  const float SL = 0.08838834764831845f * 1.4426950408889634f;  // 1/sqrt(128)*log2(e)

  for (int kb = 0; kb < nkb; ++kb) {
    const int cur = kb & 1;
    const int nxt = cur ^ 1;
    const bool more = (kb + 1 < nkb);
    if (more) {                        // async-stage next K/V tiles into LDS
      const int kn = (kb + 1) * 32;
#pragma unroll
      for (int i = 0; i < 4; ++i)
        cp_async16(&lK[nxt][rK[i] * 128 + oK[i]], &Kb[headQK + (size_t)(kn + rK[i]) * DH + oK[i]]);
#pragma unroll
      for (int i = 0; i < 4; ++i)
        cp_async16(&lV[nxt][rV[i] * 32 + oV[i]],  &Vt[headV + (size_t)rV[i] * T_SEQ + kn + oV[i]]);
    }

    // S = Q K^T  (two 16-key column tiles, K-dim = d = 128 -> 4 WMMA steps)
    v8f s0 = {}, s1 = {};
#pragma unroll
    for (int kd = 0; kd < 4; ++kd) {
      Frag16 b0, b1;
      b0.q[0] = *(const U4*)&lK[cur][l15 * 128 + kd * 32 + g * 8];
      b0.q[1] = *(const U4*)&lK[cur][l15 * 128 + kd * 32 + 16 + g * 8];
      b1.q[0] = *(const U4*)&lK[cur][(16 + l15) * 128 + kd * 32 + g * 8];
      b1.q[1] = *(const U4*)&lK[cur][(16 + l15) * 128 + kd * 32 + 16 + g * 8];
      s0 = __builtin_amdgcn_wmma_f32_16x16x32_bf16(false, qa[kd].v, false, b0.v, (short)0, s0, false, false);
      s1 = __builtin_amdgcn_wmma_f32_16x16x32_bf16(false, qa[kd].v, false, b1.v, (short)0, s1, false, false);
    }
#pragma unroll
    for (int r = 0; r < 8; ++r) { s0[r] *= SL; s1[r] *= SL; }   // log2-domain scores

    if (kb * 32 + 31 > qm0 + wave * 16) {        // diagonal blocks: causal mask
      const int k0i = kb * 32 + l15, k1i = k0i + 16;
#pragma unroll
      for (int r = 0; r < 8; ++r) {
        const int q = qm0 + wave * 16 + r + 8 * g;
        if (k0i > q) s0[r] = -1e30f;
        if (k1i > q) s1[r] = -1e30f;
      }
    }

    // online softmax: row max via ds_swizzle xor-reduction over the 16-lane group
#pragma unroll
    for (int r = 0; r < 8; ++r) {
      float rm = fmaxf(s0[r], s1[r]);
      rm = fmaxf(rm, SWZ_XOR(rm, 1));
      rm = fmaxf(rm, SWZ_XOR(rm, 2));
      rm = fmaxf(rm, SWZ_XOR(rm, 4));
      rm = fmaxf(rm, SWZ_XOR(rm, 8));
      const float mn = fmaxf(m[r], rm);
      const float alpha = exp2f(m[r] - mn);
      m[r] = mn;
      lsum[r] *= alpha;
#pragma unroll
      for (int df = 0; df < 8; ++df) o[df][r] *= alpha;
      s0[r] = exp2f(s0[r] - mn);
      s1[r] = exp2f(s1[r] - mn);
    }

    // P (C layout) -> per-wave LDS [16][32] -> reload as A fragment
#pragma unroll
    for (int r = 0; r < 8; ++r) {
      lP[wave][(r + 8 * g) * 32 + l15]      = f2bf(s0[r]);
      lP[wave][(r + 8 * g) * 32 + 16 + l15] = f2bf(s1[r]);
    }
    Frag16 pa;
    pa.q[0] = *(const U4*)&lP[wave][l15 * 32 + g * 8];
    pa.q[1] = *(const U4*)&lP[wave][l15 * 32 + 16 + g * 8];

    // O += P * V  (B from V^T tile [d][t], 8 d-tiles) ; lsum += P * ones
#pragma unroll
    for (int df = 0; df < 8; ++df) {
      Frag16 bv;
      const int vrow = df * 16 + l15;
      bv.q[0] = *(const U4*)&lV[cur][vrow * 32 + g * 8];
      bv.q[1] = *(const U4*)&lV[cur][vrow * 32 + 16 + g * 8];
      o[df] = __builtin_amdgcn_wmma_f32_16x16x32_bf16(false, pa.v, false, bv.v, (short)0, o[df], false, false);
    }
    lsum = __builtin_amdgcn_wmma_f32_16x16x32_bf16(false, pa.v, false, ones.v, (short)0, lsum, false, false);

    wait_async0();
    __syncthreads();
  }

  // normalize by row sums and stage O tile (reuse lQ), then coalesced store
#pragma unroll
  for (int r = 0; r < 8; ++r) {
    const float inv = 1.0f / lsum[r];
#pragma unroll
    for (int df = 0; df < 8; ++df)
      lQ[(wave * 16 + r + 8 * g) * 128 + df * 16 + l15] = f2bf(o[df][r] * inv);
  }
  __syncthreads();
#pragma unroll
  for (int i = 0; i < 8; ++i) {
    int c = i * 128 + tid;
    int row = c >> 4, off = (c & 15) * 8;
    *(U4*)&O[((size_t)b * T_SEQ + qm0 + row) * D_MODEL + h * DH + off] =
        *(const U4*)&lQ[row * 128 + off];
  }
}

// ---------------------------------------------------------------------------
extern "C" void kernel_launch(void* const* d_in, const int* in_sizes, int n_in,
                              void* d_out, int out_size, void* d_ws, size_t ws_size,
                              hipStream_t stream) {
  const float* x    = (const float*)d_in[0];
  const float* cosT = (const float*)d_in[1];
  const float* sinT = (const float*)d_in[2];
  const float* Wq   = (const float*)d_in[3];
  const float* Wk   = (const float*)d_in[4];
  const float* Wv   = (const float*)d_in[5];
  const float* Wo   = (const float*)d_in[6];
  float* out = (float*)d_out;

  const size_t SZ_X = (size_t)NBATCH * T_SEQ * D_MODEL;
  const size_t SZ_W = (size_t)D_MODEL * D_MODEL;

  u16* xb  = (u16*)d_ws;          // x in bf16
  u16* Wqt = xb  + SZ_X;          // transposed bf16 weights
  u16* Wkt = Wqt + SZ_W;
  u16* Wvt = Wkt + SZ_W;
  u16* Wot = Wvt + SZ_W;
  u16* Qb  = Wot + SZ_W;          // [b][h][t][128]
  u16* Kb  = Qb  + SZ_X;          // [b][h][t][128]
  u16* Vtb = Kb  + SZ_X;          // [b][h][d][T]
  u16* Ob  = Vtb + SZ_X;          // [b*t][2048]

  convx_kernel<<<dim3((unsigned)(SZ_X / 4 / 256)), dim3(256), 0, stream>>>(x, xb, (int)(SZ_X / 4));
  dim3 tg(64, 64), tb(32, 8);
  wtrans_kernel<<<tg, tb, 0, stream>>>(Wq, Wqt);
  wtrans_kernel<<<tg, tb, 0, stream>>>(Wk, Wkt);
  wtrans_kernel<<<tg, tb, 0, stream>>>(Wv, Wvt);
  wtrans_kernel<<<tg, tb, 0, stream>>>(Wo, Wot);

  dim3 pg(128, 16);
  proj_kernel<0><<<pg, 128, 0, stream>>>(xb, Wqt, cosT, sinT, Qb, nullptr);
  proj_kernel<0><<<pg, 128, 0, stream>>>(xb, Wkt, cosT, sinT, Kb, nullptr);
  proj_kernel<1><<<pg, 128, 0, stream>>>(xb, Wvt, cosT, sinT, Vtb, nullptr);

  flash_kernel<<<dim3(T_SEQ / 64, NH, NBATCH), 128, 0, stream>>>(Qb, Kb, Vtb, Ob);

  proj_kernel<2><<<pg, 128, 0, stream>>>(Ob, Wot, cosT, sinT, nullptr, out);
}
